// ContextualFlow_26843545600668
// MI455X (gfx1250) — matrix operand order repaired and checked by reading, not compiled, not run on hardware
//
#include <hip/hip_runtime.h>

// Problem constants (from the reference): f,b: [4, 64, 64, 64] fp32.
#define NB  4
#define CCH 64
#define HH  64
#define WW  64
#define LSP 4096      // H*W spatial positions / number of patches
#define KP  576       // C * 3 * 3 contraction length

// Async global->LDS staging via CDNA5 GLOBAL_LOAD_ASYNC_TO_LDS_B128 (ASYNCcnt).
// Set to 0 to fall back to global_load_b128 + ds_store_b128 staging.
#define USE_ASYNC_LDS 1

typedef _Float16 half_t;
typedef __attribute__((ext_vector_type(16))) _Float16       v16h;
typedef __attribute__((ext_vector_type(8)))  float          v8f;
typedef __attribute__((ext_vector_type(8)))  unsigned short u16x8;

union Frag {
    v16h  h;
    u16x8 u[2];
};

// ---------------------------------------------------------------------------
// Phase 1: extract 3x3 zero-padded patches.
//   P [b, l, c*9 + i*3 + j]                  (f16, K-contiguous rows)
//   PT[b, c*9 + i*3 + j, l]                  (f16, transposed copy; optional)
//   w_out[((b*L + l)*C + c)*9 + i*3 + j]     (fp32 reference output; optional)
// One thread per (b, l, c); 9 taps each.
// ---------------------------------------------------------------------------
__global__ __launch_bounds__(256)
void extract_patches_kernel(const float* __restrict__ src,
                            half_t* __restrict__ P,
                            half_t* __restrict__ PT,
                            float* __restrict__ w_out) {
    int id = blockIdx.x * blockDim.x + threadIdx.x;
    if (id >= NB * LSP * CCH) return;
    int c = id % CCH;
    int l = (id / CCH) % LSP;
    int b = id / (CCH * LSP);
    int x = l % WW;
    int y = l / WW;

    const float* sp = src + ((size_t)(b * CCH + c) * HH) * WW;
    half_t* pp = P + (size_t)(b * LSP + l) * KP + c * 9;
    half_t* tp = PT ? (PT + ((size_t)b * KP + c * 9) * LSP + l) : nullptr;
    float*  wp = w_out ? (w_out + ((size_t)(b * LSP + l) * CCH + c) * 9) : nullptr;

#pragma unroll
    for (int i = 0; i < 3; ++i) {
#pragma unroll
        for (int j = 0; j < 3; ++j) {
            int yy = y + i - 1, xx = x + j - 1;
            float v = (yy >= 0 && yy < HH && xx >= 0 && xx < WW)
                        ? sp[yy * WW + xx] : 0.0f;
            pp[i * 3 + j] = (half_t)v;
            if (tp) tp[(size_t)(i * 3 + j) * LSP] = (half_t)v;
            if (wp) wp[i * 3 + j] = v;
        }
    }
}

// ---------------------------------------------------------------------------
// Tiled WMMA GEMM, f16 in / f32 out:  C[M,N] = A[M,K] * B[N,K]^T
// (both operands K-contiguous). 256 threads = 8 wave32s arranged WM x WN;
// each wave computes a (TM*16) x (TN*16) register tile of
// v_wmma_f32_16x16x32_f16 accumulators. LDS double-buffered: one barrier
// per BK=32 step; tile k+1 is staged (async, ASYNCcnt-tracked) while tile k
// is consumed. Requires M%BM==0, N%BN==0, K%32==0.
// ---------------------------------------------------------------------------
template <int BM, int BN, int WM, int WN, int TM, int TN>
__global__ __launch_bounds__(256)
void gemm_wmma(const half_t* __restrict__ A, const half_t* __restrict__ B,
               float* __restrict__ C, int M, int N, int K) {
    static_assert(WM * WN == 8, "8 wave32s per 256-thread block");
    static_assert(WM * TM * 16 == BM, "M tiling");
    static_assert(WN * TN * 16 == BN, "N tiling");
    constexpr int BK  = 32;
    constexpr int LDT = BK + 8;                   // +8 halves: bank-conflict pad
    constexpr int AVEC = (BM * BK) / (256 * 8);   // b128 chunks per thread (A)
    constexpr int BVEC = (BN * BK) / (256 * 8);   // b128 chunks per thread (B)

    __shared__ half_t As[2][BM * LDT];
    __shared__ half_t Bs[2][BN * LDT];

    const int tid  = threadIdx.x;
    const int lane = tid & 31;
    const int wave = tid >> 5;
    const int wrow = wave / WN;
    const int wcol = wave % WN;
    const int m0   = blockIdx.y * BM;
    const int n0   = blockIdx.x * BN;

    // Fragment lane mapping (ISA 7.12.2):
    //  A 16x32 f16: lanes 0-15 row=lane, K in {0..7,16..23}; lanes 16-31 K in {8..15,24..31}
    //  B 32x16    : lanes 0-15 col=lane, K in {0..7,16..23}; lanes 16-31 K in {8..15,24..31}
    const int r16   = lane & 15;
    const int ahalf = (lane >> 4) * 8;
    const int bhalf = (lane >> 4) * 8;

    v8f acc[TM][TN] = {};

    auto stage = [&](int kt, int buf) {
        const int k0 = kt * BK;
#pragma unroll
        for (int ch = 0; ch < AVEC; ++ch) {
            int idx = (ch * 256 + tid) * 8;
            int row = idx >> 5, col = idx & 31;
            half_t* dst = &As[buf][row * LDT + col];
            const half_t* src = A + (size_t)(m0 + row) * K + k0 + col;
#if USE_ASYNC_LDS
            asm volatile("global_load_async_to_lds_b128 %0, %1, off"
                         :: "v"((unsigned)(uintptr_t)dst),
                            "v"((unsigned long long)(uintptr_t)src)
                         : "memory");
#else
            *(u16x8*)dst = *(const u16x8*)src;
#endif
        }
#pragma unroll
        for (int ch = 0; ch < BVEC; ++ch) {
            int idx = (ch * 256 + tid) * 8;
            int row = idx >> 5, col = idx & 31;
            half_t* dst = &Bs[buf][row * LDT + col];
            const half_t* src = B + (size_t)(n0 + row) * K + k0 + col;
#if USE_ASYNC_LDS
            asm volatile("global_load_async_to_lds_b128 %0, %1, off"
                         :: "v"((unsigned)(uintptr_t)dst),
                            "v"((unsigned long long)(uintptr_t)src)
                         : "memory");
#else
            *(u16x8*)dst = *(const u16x8*)src;
#endif
        }
    };

    const int nk = K / BK;
    stage(0, 0);
#if USE_ASYNC_LDS
    asm volatile("s_wait_asynccnt 0x0" ::: "memory");
#endif
    __syncthreads();

    for (int kt = 0; kt < nk; ++kt) {
        const int buf = kt & 1;
        if (kt + 1 < nk) stage(kt + 1, buf ^ 1);   // overlap next-tile staging

        Frag af[TM], bf[TN];
#pragma unroll
        for (int tm = 0; tm < TM; ++tm) {
            const half_t* p = &As[buf][(wrow * TM * 16 + tm * 16 + r16) * LDT + ahalf];
            af[tm].u[0] = *(const u16x8*)p;
            af[tm].u[1] = *(const u16x8*)(p + 16);
        }
#pragma unroll
        for (int tn = 0; tn < TN; ++tn) {
            const half_t* p = &Bs[buf][(wcol * TN * 16 + tn * 16 + r16) * LDT + bhalf];
            bf[tn].u[0] = *(const u16x8*)p;
            bf[tn].u[1] = *(const u16x8*)(p + 16);
        }
#pragma unroll
        for (int tm = 0; tm < TM; ++tm)
#pragma unroll
            for (int tn = 0; tn < TN; ++tn)
                acc[tm][tn] = __builtin_amdgcn_wmma_f32_16x16x32_f16(
                    false, af[tm].h, false, bf[tn].h,
                    (short)0, acc[tm][tn], false, false);

#if USE_ASYNC_LDS
        // Ensure this iteration's async stores into buf^1 have landed before
        // the barrier that publishes them to the other waves.
        asm volatile("s_wait_asynccnt 0x0" ::: "memory");
#endif
        __syncthreads();
    }

    // D layout: VGPR r, lanes 0-15 -> (M=r, N=lane); lanes 16-31 -> (M=8+r, N=lane-16)
    const int cn = lane & 15;
    const int mb = (lane >> 4) * 8;
#pragma unroll
    for (int tm = 0; tm < TM; ++tm)
#pragma unroll
        for (int tn = 0; tn < TN; ++tn) {
            int gm = m0 + wrow * TM * 16 + tm * 16 + mb;
            int gn = n0 + wcol * TN * 16 + tn * 16 + cn;
#pragma unroll
            for (int vr = 0; vr < 8; ++vr)
                C[(size_t)(gm + vr) * N + gn] = acc[tm][tn][vr];
        }
}

// ---------------------------------------------------------------------------
// Row-wise scaled softmax: attn[p, l] = softmax_l(10 * S[p, l]), f16 output.
// One 256-thread block per row of 4096; row cached in 16 registers so the
// 64 MB score matrix is read exactly once.
// ---------------------------------------------------------------------------
__global__ __launch_bounds__(256)
void softmax_rows(const float* __restrict__ S, half_t* __restrict__ At) {
    __shared__ float red[256];
    const int row = blockIdx.x;
    const int t   = threadIdx.x;
    const float* sp = S + (size_t)row * LSP;

    float v[16];
    float m = -1e30f;
#pragma unroll
    for (int j = 0; j < 16; ++j) {
        v[j] = 10.0f * sp[t + j * 256];
        m = fmaxf(m, v[j]);
    }
    red[t] = m; __syncthreads();
    for (int s = 128; s > 0; s >>= 1) {
        if (t < s) red[t] = fmaxf(red[t], red[t + s]);
        __syncthreads();
    }
    m = red[0];
    __syncthreads();

    float sum = 0.0f;
#pragma unroll
    for (int j = 0; j < 16; ++j) {
        v[j] = __expf(v[j] - m);
        sum += v[j];
    }
    red[t] = sum; __syncthreads();
    for (int s = 128; s > 0; s >>= 1) {
        if (t < s) red[t] += red[t + s];
        __syncthreads();
    }
    const float inv = 1.0f / red[0];

    half_t* ap = At + (size_t)row * LSP;
#pragma unroll
    for (int j = 0; j < 16; ++j)
        ap[t + j * 256] = (half_t)(v[j] * inv);
}

// ---------------------------------------------------------------------------
// Fold (col2im): y[c,h,w] = (1/9) * sum_{i,j} U[(h+1-i)*W + (w+1-j), c*9+3i+j]
// ---------------------------------------------------------------------------
__global__ __launch_bounds__(256)
void fold_kernel(const float* __restrict__ U, float* __restrict__ y) {
    int id = blockIdx.x * blockDim.x + threadIdx.x;
    if (id >= CCH * HH * WW) return;
    int x = id % WW;
    int h = (id / WW) % HH;
    int c = id / (WW * HH);
    float s = 0.0f;
#pragma unroll
    for (int i = 0; i < 3; ++i) {
#pragma unroll
        for (int j = 0; j < 3; ++j) {
            int py = h + 1 - i, px = x + 1 - j;
            if (py >= 0 && py < HH && px >= 0 && px < WW)
                s += U[(size_t)(py * WW + px) * KP + c * 9 + i * 3 + j];
        }
    }
    y[id] = s * (1.0f / 9.0f);
}

// ---------------------------------------------------------------------------
extern "C" void kernel_launch(void* const* d_in, const int* in_sizes, int n_in,
                              void* d_out, int out_size, void* d_ws, size_t ws_size,
                              hipStream_t stream) {
    const float* f = (const float*)d_in[0];
    const float* b = (const float*)d_in[1];

    float* y     = (float*)d_out;                            // [B, C, H, W]
    float* w_out = y + (size_t)NB * CCH * HH * WW;           // [B, L*C, 3, 3]

    // Workspace carve-up (256B aligned). Total ~162 MB.
    char*  ws  = (char*)d_ws;
    size_t off = 0;
    auto carve = [&](size_t bytes) -> void* {
        void* p = ws + off;
        off += (bytes + 255) & ~(size_t)255;
        return p;
    };
    half_t* Pf  = (half_t*)carve((size_t)NB * LSP * KP * sizeof(half_t)); // 18.9 MB
    half_t* Pb  = (half_t*)carve((size_t)NB * LSP * KP * sizeof(half_t)); // 18.9 MB
    half_t* PbT = (half_t*)carve((size_t)NB * KP * LSP * sizeof(half_t)); // 18.9 MB
    float*  S   = (float*) carve((size_t)LSP * LSP * sizeof(float));      // 64 MB (per-batch reuse)
    half_t* At  = (half_t*)carve((size_t)LSP * LSP * sizeof(half_t));     // 32 MB (per-batch reuse)
    float*  U   = (float*) carve((size_t)LSP * KP  * sizeof(float));      //  9.4 MB (per-batch reuse)

    const int nExtract = NB * LSP * CCH;
    extract_patches_kernel<<<(nExtract + 255) / 256, 256, 0, stream>>>(f, Pf, nullptr, nullptr);
    extract_patches_kernel<<<(nExtract + 255) / 256, 256, 0, stream>>>(b, Pb, PbT, w_out);

    dim3 g1(LSP / 128, LSP / 128);  // scores: 32 x 32 blocks, M=N=4096, K=576
    dim3 g2(KP / 64,   LSP / 128);  // U: 9 x 32 blocks, M=4096, N=576, K=4096
    const int nFold = CCH * HH * WW;

    for (int bb = 0; bb < NB; ++bb) {
        const half_t* Pfb  = Pf  + (size_t)bb * LSP * KP;
        const half_t* Pbb  = Pb  + (size_t)bb * LSP * KP;
        const half_t* PbTb = PbT + (size_t)bb * KP * LSP;
        // scores = Pf * Pb^T        (A=[L,576], B=[L,576] K-contiguous)
        gemm_wmma<128, 128, 2, 4, 4, 2><<<g1, 256, 0, stream>>>(Pfb, Pbb, S, LSP, LSP, KP);
        // attn = softmax(10 * scores) over l, emitted as f16
        softmax_rows<<<LSP, 256, 0, stream>>>(S, At);
        // U = attn * Pb = attn * PbT^T   (A=[L,L], B=PbT [576,L] K-contiguous)
        gemm_wmma<128, 64, 4, 2, 2, 2><<<g2, 256, 0, stream>>>(At, PbTb, U, LSP, KP, LSP);
        // fold + /9
        fold_kernel<<<(nFold + 255) / 256, 256, 0, stream>>>(U, y + (size_t)bb * CCH * HH * WW);
    }
}
